// Temper_35734127903246
// MI455X (gfx1250) — compile-verified, matrix operand order checked by reading, not run on hardware
//
#include <hip/hip_runtime.h>

// ---------------------------------------------------------------------------
// MoE routed 2-layer MLP, bf16 WMMA (V_WMMA_F32_16X16X32_BF16), gfx1250.
//
// Roofline: 17.2 GFLOP routed compute vs 256 MB mandatory HBM traffic
// (x read + out write) => memory-bound at ~11us @ 23.3 TB/s on the bf16
// matrix pipe (fp32 WMMA 16x16x4 would be compute-bound).
// This revision amortizes the register-resident weight fragments over
// ITER=4 consecutive 64-sample blocks per workgroup, cutting L2 weight
// traffic from 256 MB to 64 MB so it stays well under the HBM roof.
// ---------------------------------------------------------------------------

typedef __attribute__((ext_vector_type(16))) __bf16 v16bf;
typedef __attribute__((ext_vector_type(8)))  float  v8f;

#define DD      128        // feature dim
#define NE      3          // experts
#define NB      262144     // batch
#define MBLK    64         // samples per block-iteration (4 M-tiles of 16)
#define ITER    4          // m-blocks processed per workgroup
#define XSTRIDE 136        // bf16 row stride in LDS: 272B = 68 dwords, 68%64=4
                           // -> consecutive rows shift 4 banks, conflict-free b128

union FragU { v16bf v; uint4 q[2]; };

// Build a 16x32 bf16 WMMA fragment chunk for (K-step t, lane-half h):
// lane holds K in {32t+8h .. +7} and {32t+16+8h .. +7}  -> two 16B loads.
__device__ __forceinline__ v16bf load_frag(const __bf16* row, int t, int half) {
    FragU f;
    f.q[0] = *(const uint4*)(row + t * 32 + half * 8);
    f.q[1] = *(const uint4*)(row + t * 32 + 16 + half * 8);
    return f.v;
}

__global__ void zero_counts_kernel(int* counts) {
    if (threadIdx.x < NE) counts[threadIdx.x] = 0;
}

__global__ void bucket_kernel(const int* __restrict__ ops,
                              int* counts, int* __restrict__ idxs) {
    int i = blockIdx.x * blockDim.x + threadIdx.x;
    if (i < NB) {
        int e = ops[i];
        int p = atomicAdd(&counts[e], 1);
        idxs[e * NB + p] = i;   // order nondeterministic, output invariant
    }
}

__global__ void wconv_kernel(const float* __restrict__ W1,
                             const float* __restrict__ W2,
                             __bf16* __restrict__ Wb1,
                             __bf16* __restrict__ Wb2) {
    int i = blockIdx.x * blockDim.x + threadIdx.x;
    if (i < NE * DD * DD) {
        Wb1[i] = (__bf16)W1[i];
        Wb2[i] = (__bf16)W2[i];
    }
}

__global__ __launch_bounds__(256)
void moe_gemm_kernel(const float* __restrict__ x,
                     const __bf16* __restrict__ Wb1,
                     const __bf16* __restrict__ Wb2,
                     const float* __restrict__ b1,
                     const float* __restrict__ b2,
                     const float* __restrict__ fresh,
                     const int* __restrict__ counts,
                     const int* __restrict__ idxs,
                     float* __restrict__ out) {
    __shared__ __bf16 xt[MBLK][XSTRIDE];   // gathered bf16 inputs
    __shared__ __bf16 ht[MBLK][XSTRIDE];   // layer-1 activations
    __shared__ int    srow[MBLK];          // gathered sample indices

    const int e     = blockIdx.y;
    const int count = counts[e];
    const int blk0  = blockIdx.x * ITER;
    if (blk0 * MBLK >= count) return;      // block-uniform exit

    const int tid  = threadIdx.x;
    const int wave = tid >> 5;             // 8 waves, wave owns N-tile
    const int lane = tid & 31;
    const int half = lane >> 4;
    const int l    = lane & 15;
    const int n    = wave * 16 + l;        // this lane's output column

    // Weight fragments stay in VGPRs for all ITER blocks (64 VGPRs).
    v16bf w1f[4], w2f[4];
    const __bf16* w1row = Wb1 + ((size_t)e * DD + n) * DD;
    const __bf16* w2row = Wb2 + ((size_t)e * DD + n) * DD;
#pragma unroll
    for (int t = 0; t < 4; ++t) {
        w1f[t] = load_frag(w1row, t, half);
        w2f[t] = load_frag(w2row, t, half);
    }
    const float bias1 = b1[e * DD + n];
    const float bias2 = b2[e * DD + n];
    const float boost = (fresh[e] > 0.0f) ? 1.01f : 1.0f;

    for (int it = 0; it < ITER; ++it) {
        const int mb = blk0 + it;
        if (mb * MBLK >= count) break;     // uniform: count/mb are scalar

        __syncthreads();   // protect srow/xt from previous iteration readers

        if (tid < MBLK) {
            int p = mb * MBLK + tid;
            srow[tid] = idxs[e * NB + min(p, count - 1)];  // clamp padding
        }
        __syncthreads();

        // Gather + fp32->bf16 convert the x tile (MBLK x 128), float4 chunks.
        for (int c = tid; c < MBLK * (DD / 4); c += 256) {
            int r  = c >> 5;        // row within block
            int c4 = c & 31;        // float4 chunk within row
            const float4 v = *(const float4*)(x + (size_t)srow[r] * DD + c4 * 4);
            __bf16* dst = &xt[r][c4 * 4];
            dst[0] = (__bf16)v.x; dst[1] = (__bf16)v.y;
            dst[2] = (__bf16)v.z; dst[3] = (__bf16)v.w;
        }
        __syncthreads();   // xt ready

        // ---- Layer 1: h = relu(x @ W1^T + b1), bf16 -> LDS ----
#pragma unroll
        for (int m = 0; m < MBLK / 16; ++m) {
            v8f acc = {};
#pragma unroll
            for (int t = 0; t < 4; ++t) {
                v16bf a = load_frag(&xt[m * 16 + l][0], t, half);
                acc = __builtin_amdgcn_wmma_f32_16x16x32_bf16(
                    false, a, false, w1f[t], (short)0, acc, false, false);
            }
#pragma unroll
            for (int r = 0; r < 8; ++r) {       // D layout: M = r + 8*half
                float v = acc[r] + bias1;
                v = v > 0.0f ? v : 0.0f;
                ht[m * 16 + r + half * 8][n] = (__bf16)v;
            }
        }
        __syncthreads();   // ht ready (all waves contributed all columns)

        // ---- Layer 2: out = boost * relu(h @ W2^T + b2), scatter fp32 ----
#pragma unroll
        for (int m = 0; m < MBLK / 16; ++m) {
            v8f acc = {};
#pragma unroll
            for (int t = 0; t < 4; ++t) {
                v16bf a = load_frag(&ht[m * 16 + l][0], t, half);
                acc = __builtin_amdgcn_wmma_f32_16x16x32_bf16(
                    false, a, false, w2f[t], (short)0, acc, false, false);
            }
#pragma unroll
            for (int r = 0; r < 8; ++r) {
                int   rr = m * 16 + r + half * 8;
                int   p  = mb * MBLK + rr;
                float v  = acc[r] + bias2;
                v = v > 0.0f ? v : 0.0f;
                v *= boost;
                if (p < count)                   // drop padded rows
                    out[(size_t)srow[rr] * DD + n] = v;
            }
        }
    }
}

extern "C" void kernel_launch(void* const* d_in, const int* in_sizes, int n_in,
                              void* d_out, int out_size, void* d_ws, size_t ws_size,
                              hipStream_t stream) {
    (void)in_sizes; (void)n_in; (void)out_size; (void)ws_size;
    const float* x     = (const float*)d_in[0];
    const float* W1    = (const float*)d_in[1];
    const float* b1    = (const float*)d_in[2];
    const float* W2    = (const float*)d_in[3];
    const float* b2    = (const float*)d_in[4];
    const float* fresh = (const float*)d_in[5];
    const int*   ops   = (const int*)d_in[6];
    float*       out   = (float*)d_out;

    // Workspace layout: counts | index lists | bf16 W1 | bf16 W2  (~3.4 MB)
    char*   ws     = (char*)d_ws;
    int*    counts = (int*)ws;
    int*    idxs   = (int*)(ws + 256);
    __bf16* Wb1    = (__bf16*)(ws + 256 + (size_t)NE * NB * sizeof(int));
    __bf16* Wb2    = Wb1 + (size_t)NE * DD * DD;

    zero_counts_kernel<<<1, 32, 0, stream>>>(counts);
    bucket_kernel<<<NB / 256, 256, 0, stream>>>(ops, counts, idxs);
    wconv_kernel<<<(NE * DD * DD + 255) / 256, 256, 0, stream>>>(W1, W2, Wb1, Wb2);

    const int nblocks = NB / MBLK;                      // 4096
    dim3 grid((nblocks + ITER - 1) / ITER, NE);         // 1024 x 3
    moe_gemm_kernel<<<grid, 256, 0, stream>>>(x, Wb1, Wb2, b1, b2, fresh,
                                              counts, idxs, out);
}
